// MultiLevelPredLayer_85796266705063
// MI455X (gfx1250) — matrix operand order repaired and checked by reading, not compiled, not run on hardware
//
#include <hip/hip_runtime.h>

typedef __attribute__((ext_vector_type(16))) _Float16 v16h;
typedef __attribute__((ext_vector_type(8)))  _Float16 v8h;
typedef __attribute__((ext_vector_type(8)))  float    v8f;

#define CDIM      256
#define OUTC      149
#define TOTAL_M   8400
#define BATCH     16
#define LDS_PITCH 264   // halfs; 528B row stride => conflict-free B-frag ds_load_b128

struct HeadParams {
  const float* cls_feat[3];
  const float* reg_feat[3];
  const float* obj_w[3];
  const float* obj_b[3];
  const float* cls_w[3];
  const float* cls_b[3];
  const float* reg_w[3];
  const float* reg_b[3];
  float* out;
};

// Pre-packed weights in A-fragment order: [zl(6)][ot(5)][ks(8)][lane(32)][16 halfs]
// zl = z*3 + lvl; rows 65..79 of the obj+reg group are zero-padded.
__device__ __attribute__((aligned(32))) _Float16 g_packedW[6 * 5 * 8 * 32 * 16];
__device__ __attribute__((aligned(32))) float    g_packedB[6 * 80];

// grid = (6, 5); block = 256 = 8 ks-groups x 32 lanes. Converts fp32 weights -> f16
// fragments once per launch; also packs the bias table.
__global__ __launch_bounds__(256) void pack_weights_kernel(HeadParams p) {
  const int zl  = blockIdx.x;      // 0..5
  const int ot  = blockIdx.y;      // 0..4
  const int z   = zl / 3;
  const int lvl = zl % 3;
  const int tid  = threadIdx.x;
  const int ks   = tid >> 5;
  const int lane = tid & 31;
  const int mrow = lane & 15;
  const int half = lane >> 4;
  const int wr   = ot * 16 + mrow;

  const float* wrow = nullptr;
  if (z == 0)        wrow = p.cls_w[lvl] + (size_t)wr * CDIM;
  else if (wr == 0)  wrow = p.obj_w[lvl];
  else if (wr <= 64) wrow = p.reg_w[lvl] + (size_t)(wr - 1) * CDIM;

  const int kb = ks * 32 + half * 8;
  _Float16* dst = g_packedW + ((((size_t)zl * 5 + ot) * 8 + ks) * 32 + lane) * 16;
  if (wrow) {
#pragma unroll
    for (int j = 0; j < 8; ++j) {
      dst[j]     = (_Float16)wrow[kb + j];
      dst[8 + j] = (_Float16)wrow[kb + 16 + j];
    }
  } else {
#pragma unroll
    for (int j = 0; j < 16; ++j) dst[j] = (_Float16)0.0f;
  }

  if (ot == 0 && tid < 80) {
    float bv = 0.0f;
    if (z == 0)         bv = p.cls_b[lvl][tid];
    else if (tid == 0)  bv = p.obj_b[lvl][0];
    else if (tid <= 64) bv = p.reg_b[lvl][tid - 1];
    g_packedB[zl * 80 + tid] = bv;
  }
}

// grid = (8400, 1, 2): x enumerates (batch, level, 16-anchor tile); z: 0=cls, 1=obj+reg
// block = 64 threads = 2 wave32; each wave covers output tiles {wave, wave+2, ...} of 5.
__global__ __launch_bounds__(64) void head_gemm_kernel(HeadParams p) {
  __shared__ _Float16 ldsB[16 * LDS_PITCH];

  const int tid  = threadIdx.x;
  const int lane = tid & 31;
  const int wave = tid >> 5;

  int bx = blockIdx.x;
  int b  = bx / 525;          // 525 = 400 + 100 + 25 anchor tiles per batch item
  int t  = bx - b * 525;
  int lvl, tile, Mlvl, moff;
  if (t < 400)      { lvl = 0; tile = t;       Mlvl = 6400; moff = 0;    }
  else if (t < 500) { lvl = 1; tile = t - 400; Mlvl = 1600; moff = 6400; }
  else              { lvl = 2; tile = t - 500; Mlvl = 400;  moff = 8000; }
  const int z  = blockIdx.z;
  const int zl = z * 3 + lvl;
  const int m0 = tile * 16;

  const float* feat = (z == 0 ? p.cls_feat[lvl] : p.reg_feat[lvl])
                    + (size_t)b * CDIM * Mlvl + m0;

  // ---- stage 16 anchors x 256 channels into LDS as f16 (packed dword stores) ----
  for (int it = 0; it < 32; ++it) {
    int idx = it * 64 + tid;           // (anchor, channel-pair)
    int n   = idx & 15;
    int c2  = idx >> 4;                // 0..127
    float f0 = feat[(size_t)(2 * c2 + 0) * Mlvl + n];
    float f1 = feat[(size_t)(2 * c2 + 1) * Mlvl + n];
    union { _Float16 h[2]; unsigned int u; } pk;
    pk.h[0] = (_Float16)f0;
    pk.h[1] = (_Float16)f1;
    *(unsigned int*)&ldsB[n * LDS_PITCH + 2 * c2] = pk.u;
  }
  __syncthreads();

  const int n    = lane & 15;   // B column / C column (anchor within tile)
  const int half = lane >> 4;   // K-half selector

  // ---- B fragments for all 8 K-steps, kept in registers for all out-tiles ----
  v16h bfrag[8];
#pragma unroll
  for (int ks = 0; ks < 8; ++ks) {
    const v8h* bp = (const v8h*)&ldsB[n * LDS_PITCH + ks * 32 + half * 8];
    v8h b0 = bp[0];
    v8h b1 = bp[2];
    bfrag[ks] = __builtin_shufflevector(b0, b1, 0,1,2,3,4,5,6,7,8,9,10,11,12,13,14,15);
  }

  const int mglob = moff + m0 + n;
  float* orow = p.out + ((size_t)b * TOTAL_M + mglob) * OUTC;
  const _Float16* wbase = g_packedW + (size_t)zl * 5 * 8 * 32 * 16 + lane * 16;
  const float*    pb    = g_packedB + zl * 80;

  for (int ot = wave; ot < 5; ot += 2) {
    const _Float16* wt = wbase + (size_t)ot * 8 * 32 * 16;
    v8f acc = {};
#pragma unroll
    for (int ks = 0; ks < 8; ++ks) {
      v16h afrag = *(const v16h*)(wt + (size_t)ks * 32 * 16);  // 32B aligned, coalesced
      acc = __builtin_amdgcn_wmma_f32_16x16x32_f16(
                false, afrag, false, bfrag[ks], (short)0, acc, false, false);
    }

    // Bias rows for this lane's 8 C elements are contiguous: one 32B load.
    v8f bias = *(const v8f*)(pb + ot * 16 + 8 * half);

    // C layout: VGPR r, lane l -> row = r + 8*half, col(anchor) = n
    if (z == 0) {
#pragma unroll
      for (int r = 0; r < 8; ++r) {
        int wrr = ot * 16 + r + 8 * half;
        orow[1 + wrr] = acc[r] + bias[r];            // cls cols 1..80
      }
    } else {
#pragma unroll
      for (int r = 0; r < 8; ++r) {
        int wrr = ot * 16 + r + 8 * half;
        int col = (wrr == 0) ? 0 : 80 + wrr;         // obj col 0, reg cols 81..144
        float v = acc[r] + bias[r];
        if (wrr <= 64) orow[col] = v;                // rows 65..79 are padding
      }
    }
  }
}

// DFL decode + box: one thread per (batch, anchor). proj = linspace(0,16,16) => step 16/15.
__global__ __launch_bounds__(256) void dfl_box_kernel(float* out) {
  int gid = blockIdx.x * 256 + threadIdx.x;
  if (gid >= BATCH * TOTAL_M) return;
  int b  = gid / TOTAL_M;
  int mg = gid - b * TOTAL_M;
  int m, W, stride;
  if (mg < 6400)      { m = mg;        W = 80; stride = 8;  }
  else if (mg < 8000) { m = mg - 6400; W = 40; stride = 16; }
  else                { m = mg - 8000; W = 20; stride = 32; }
  int h = m / W, w = m - h * W;
  float ax = (w + 0.5f) * (float)stride;
  float ay = (h + 0.5f) * (float)stride;

  float* row = out + (size_t)gid * OUTC;
  float d[4];
#pragma unroll
  for (int f = 0; f < 4; ++f) {
    float v[16];
#pragma unroll
    for (int j = 0; j < 16; ++j) v[j] = row[81 + f * 16 + j];
    float mx = v[0];
#pragma unroll
    for (int j = 1; j < 16; ++j) mx = fmaxf(mx, v[j]);
    const float scale = 16.0f / 15.0f;
    float s = 0.0f, e = 0.0f;
#pragma unroll
    for (int j = 0; j < 16; ++j) {
      float ev = __expf(v[j] - mx);
      s += ev;
      e += ev * ((float)j * scale);
    }
    d[f] = e / s;
  }
  row[145] = ax - d[0] * (float)stride;
  row[146] = ay - d[1] * (float)stride;
  row[147] = ax + d[2] * (float)stride;
  row[148] = ay + d[3] * (float)stride;
}

extern "C" void kernel_launch(void* const* d_in, const int* in_sizes, int n_in,
                              void* d_out, int out_size, void* d_ws, size_t ws_size,
                              hipStream_t stream) {
  (void)in_sizes; (void)n_in; (void)d_ws; (void)ws_size; (void)out_size;
  HeadParams p;
  for (int l = 0; l < 3; ++l) {
    p.cls_feat[l] = (const float*)d_in[2 * l + 0];
    p.reg_feat[l] = (const float*)d_in[2 * l + 1];
    p.obj_w[l] = (const float*)d_in[6 + 6 * l + 0];
    p.obj_b[l] = (const float*)d_in[6 + 6 * l + 1];
    p.cls_w[l] = (const float*)d_in[6 + 6 * l + 2];
    p.cls_b[l] = (const float*)d_in[6 + 6 * l + 3];
    p.reg_w[l] = (const float*)d_in[6 + 6 * l + 4];
    p.reg_b[l] = (const float*)d_in[6 + 6 * l + 5];
  }
  p.out = (float*)d_out;

  pack_weights_kernel<<<dim3(6, 5), dim3(256), 0, stream>>>(p);
  head_gemm_kernel<<<dim3(BATCH * 525, 1, 2), dim3(64), 0, stream>>>(p);

  int total = BATCH * TOTAL_M;
  dfl_box_kernel<<<dim3((total + 255) / 256), dim3(256), 0, stream>>>(p.out);
}
